// BahdanauDecoder_52656299049482
// MI455X (gfx1250) — compile-verified, matrix operand order checked by reading, not run on hardware
//
#include <hip/hip_runtime.h>
#include <hip/hip_bf16.h>

// CDNA5 / gfx1250: wave32, WMMA 16x16x32 bf16 path.
typedef __attribute__((ext_vector_type(16))) __bf16 v16bf;
typedef __attribute__((ext_vector_type(8)))  float  v8f;

#define H_DIM 1024
#define S_DIM 1024
#define B_DIM 32
#define V_DIM 50257

// Fast transcendentals: CDNA5 has V_TANH_F32 as a TRANS op.
__device__ __forceinline__ float fast_tanh(float x) {
#if __has_builtin(__builtin_amdgcn_tanhf)
  return __builtin_amdgcn_tanhf(x);
#else
  float t = __expf(-2.f * fabsf(x));
  float r = (1.f - t) / (1.f + t);
  return copysignf(r, x);
#endif
}
__device__ __forceinline__ float fast_sigmoid(float x) {
  return 1.f / (1.f + __expf(-x));
}

// ---------------------------------------------------------------------------
// WMMA fragment loaders. ISA 16-bit A (16x32 MxK) layout, wave32:
//   lanes 0-15: row = lane,     kbase = k0
//   lanes16-31: row = lane-16,  kbase = k0+8
//   vector elem e -> K = kbase + e + (e>=8 ? 8 : 0)
// B (32x16 KxN) mirrors with lane <-> N column; all GEMMs are X @ W^T with W
// stored [N,K] row-major, so B fragments load W rows directly.
// ---------------------------------------------------------------------------
__device__ __forceinline__ v16bf frag_global(const float* __restrict__ p,
                                             long ld, int row0, int k0, int lane) {
  int r = row0 + (lane & 15);
  const float* rp = p + (long)r * ld + k0 + ((lane >> 4) << 3);
  __builtin_prefetch(rp + 32, 0, 3);   // next k-slab, same row (speculative)
  v16bf f;
#pragma unroll
  for (int e = 0; e < 16; ++e) {
    int kk = e + ((e >= 8) ? 8 : 0);
    f[e] = (__bf16)rp[kk];
  }
  return f;
}

__device__ __forceinline__ v16bf frag_global_guard(const float* __restrict__ p,
                                                   long ld, int row0, int k0,
                                                   int lane, int rowMax) {
  int r = row0 + (lane & 15);
  v16bf f;
  if (r < rowMax) {
    const float* rp = p + (long)r * ld + k0 + ((lane >> 4) << 3);
#pragma unroll
    for (int e = 0; e < 16; ++e) {
      int kk = e + ((e >= 8) ? 8 : 0);
      f[e] = (__bf16)rp[kk];
    }
  } else {
#pragma unroll
    for (int e = 0; e < 16; ++e) f[e] = (__bf16)0.f;
  }
  return f;
}

// LDS-staged B fragments: slab layout [n_local][16 x bf16-pair] (k-major)
__device__ __forceinline__ v16bf frag_lds(const unsigned* sBuf, int n0, int lane) {
  const __bf16* rp = (const __bf16*)(sBuf + (n0 + (lane & 15)) * 16) +
                     ((lane >> 4) << 3);
  v16bf f;
#pragma unroll
  for (int e = 0; e < 16; ++e) {
    int kk = e + ((e >= 8) ? 8 : 0);
    f[e] = rp[kk];
  }
  return f;
}

// Stage a 128-row x 32-k slab of W (f32) into LDS as packed bf16x2.
// 1024 float4 chunks / 256 threads = 4 per thread; 8 threads cover one row
// (128B coalesced global_load_b128), LDS store as uint2 (ds_store_b64).
__device__ __forceinline__ void stage_slab(unsigned* dst,
                                           const float* __restrict__ W,
                                           int n0, int k0, int K, int tid) {
#pragma unroll
  for (int it = 0; it < 4; ++it) {
    int c = tid + it * 256;
    int nl = c >> 3, kq = c & 7;
    const float4* wp = (const float4*)(W + (long)(n0 + nl) * K + k0) + kq;
    float4 f = *wp;
    union { __bf16 h[4]; uint2 u; } pk;
    pk.h[0] = (__bf16)f.x;
    pk.h[1] = (__bf16)f.y;
    pk.h[2] = (__bf16)f.z;
    pk.h[3] = (__bf16)f.w;
    ((uint2*)dst)[nl * 8 + kq] = pk.u;
  }
}

// ---------------------------------------------------------------------------
// Fused attention-score kernel:
//   scores[m] = sum_n v[n] * tanh( q[m/rowsPerQ, n] + (X @ W^T)[m, n] )
// (scalar softmax bias dropped: it cancels in softmax)
// Block = 256 thr (8 waves), 256 rows/block; wave w owns rows [32w, 32w+32)
// as 2 M-tiles (B fragment reused across both -> 16 WMMAs per k-step).
// N in 128-wide chunks; W k-slab double-buffered in LDS as bf16, one barrier
// per k-step; B fragment double-buffered in registers (ds latency overlap).
// Per-lane partial scores accumulate in registers across all N-chunks; a
// single shfl_xor reduction at the end writes rows straight to global.
// ---------------------------------------------------------------------------
__global__ __launch_bounds__(256) void fused_score_kernel(
    const float* __restrict__ X, const float* __restrict__ W,
    const float* __restrict__ q, const float* __restrict__ vvec,
    float* __restrict__ scores, int Mtotal, int rowsPerQ) {
  const int K = H_DIM, N = H_DIM;
  __shared__ unsigned sB[2][128 * 16];
  const int tid = threadIdx.x, lane = tid & 31, wave = tid >> 5;
  const int rowBlk = blockIdx.x * 256;
  const int myRow = rowBlk + wave * 32;        // 2 M-tiles per wave
  const bool rowOk = (myRow + 31) < Mtotal;    // wave-uniform
  const bool qUni = rowOk && ((myRow / rowsPerQ) == ((myRow + 31) / rowsPerQ));
  const int qbBase = myRow / rowsPerQ;

  float partial[2][8];
#pragma unroll
  for (int mt = 0; mt < 2; ++mt)
#pragma unroll
    for (int r = 0; r < 8; ++r) partial[mt][r] = 0.f;

  for (int n0 = 0; n0 < N; n0 += 128) {
    v8f acc[2][8];
#pragma unroll
    for (int mt = 0; mt < 2; ++mt)
#pragma unroll
      for (int t = 0; t < 8; ++t)
#pragma unroll
        for (int e = 0; e < 8; ++e) acc[mt][t][e] = 0.f;

    stage_slab(sB[0], W, n0, 0, K, tid);
    __syncthreads();
    int buf = 0;
    for (int k0 = 0; k0 < K; k0 += 32) {
      if (rowOk) {
        v16bf a0 = frag_global(X, K, myRow, k0, lane);
        v16bf a1 = frag_global(X, K, myRow + 16, k0, lane);
        v16bf bcur = frag_lds(sB[buf], 0, lane);
#pragma unroll
        for (int t = 0; t < 8; ++t) {
          v16bf bnxt;
          if (t < 7) bnxt = frag_lds(sB[buf], (t + 1) * 16, lane);
          acc[0][t] = __builtin_amdgcn_wmma_f32_16x16x32_bf16(
              false, a0, false, bcur, (short)0, acc[0][t], false, false);
          acc[1][t] = __builtin_amdgcn_wmma_f32_16x16x32_bf16(
              false, a1, false, bcur, (short)0, acc[1][t], false, false);
          if (t < 7) bcur = bnxt;
        }
      }
      if (k0 + 32 < K) stage_slab(sB[buf ^ 1], W, n0, k0 + 32, K, tid);
      __syncthreads();
      buf ^= 1;
    }

    if (rowOk) {
#pragma unroll
      for (int t = 0; t < 8; ++t) {
        int n = n0 + t * 16 + (lane & 15);
        float vn = vvec[n];
        float qv = qUni ? q[(long)qbBase * H_DIM + n] : 0.f;
#pragma unroll
        for (int mt = 0; mt < 2; ++mt)
#pragma unroll
          for (int r = 0; r < 8; ++r) {
            int m = mt * 16 + ((lane < 16) ? r : (r + 8));
            float qq = qUni ? qv : q[(long)((myRow + m) / rowsPerQ) * H_DIM + n];
            partial[mt][r] += vn * fast_tanh(acc[mt][t][r] + qq);
          }
      }
    }
  }

  // Final cross-lane reduction: lanes 0..15 hold row m = mt*16+r pieces,
  // lanes 16..31 hold row m = mt*16+r+8 pieces. After the 16-lane tree,
  // lane 0 / lane 16 write the two rows directly to global.
  if (rowOk) {
#pragma unroll
    for (int mt = 0; mt < 2; ++mt)
#pragma unroll
      for (int r = 0; r < 8; ++r) {
        float val = partial[mt][r];
        val += __shfl_xor(val, 1, 32);
        val += __shfl_xor(val, 2, 32);
        val += __shfl_xor(val, 4, 32);
        val += __shfl_xor(val, 8, 32);
        if ((lane & 15) == 0)
          scores[myRow + mt * 16 + r + ((lane >> 4) << 3)] = val;
      }
  }
}

// ---------------------------------------------------------------------------
// Generic M=32 GEMM with K-split: C = A @ [W1|W2]^T + b1 + b2
// Block = 128 thr (4 waves). Wave w: M-tile (w&1)*16, N-cols (w>>1)*64..+64.
// N-tail guarded (classifier N=50257). B fragments batched before WMMAs.
// ---------------------------------------------------------------------------
__global__ __launch_bounds__(128) void gemm32_kernel(
    const float* __restrict__ A, int lda,
    const float* __restrict__ W1, int K1,
    const float* __restrict__ W2, int K2,
    const float* __restrict__ b1, const float* __restrict__ b2,
    float* __restrict__ C, int N) {
  const int tid = threadIdx.x, lane = tid & 31, wave = tid >> 5;
  const int n0 = blockIdx.x * 128 + (wave >> 1) * 64;
  const int m0 = (wave & 1) * 16;
  v8f acc[4];
#pragma unroll
  for (int t = 0; t < 4; ++t)
#pragma unroll
    for (int e = 0; e < 8; ++e) acc[t][e] = 0.f;

  const int K = K1 + K2;
  for (int k0 = 0; k0 < K; k0 += 32) {
    const float* Wp = W1; int kw = k0; int ldw = K1;
    if (k0 >= K1) { Wp = W2; kw = k0 - K1; ldw = K2; }
    v16bf a = frag_global(A, lda, m0, k0, lane);
    v16bf bfr[4];
#pragma unroll
    for (int t = 0; t < 4; ++t)
      bfr[t] = frag_global_guard(Wp, ldw, n0 + t * 16, kw, lane, N);
#pragma unroll
    for (int t = 0; t < 4; ++t)
      acc[t] = __builtin_amdgcn_wmma_f32_16x16x32_bf16(
          false, a, false, bfr[t], (short)0, acc[t], false, false);
  }

#pragma unroll
  for (int t = 0; t < 4; ++t) {
    int n = n0 + t * 16 + (lane & 15);
    if (n < N) {
      float bias = (b1 ? b1[n] : 0.f) + (b2 ? b2[n] : 0.f);
#pragma unroll
      for (int r = 0; r < 8; ++r) {
        int m = m0 + ((lane < 16) ? r : (r + 8));
        C[(long)m * N + n] = acc[t][r] + bias;
      }
    }
  }
}

// ---------------------------------------------------------------------------
// Softmax over S for del (blocks 0..31) and add (blocks 32..63)
// ---------------------------------------------------------------------------
__global__ __launch_bounds__(256) void softmaxS_kernel(
    const float* __restrict__ sDel, const float* __restrict__ sAdd,
    float* __restrict__ awDel, float* __restrict__ awAdd) {
  const int b = blockIdx.x & 31;
  const bool isAdd = blockIdx.x >= 32;
  const float* s = (isAdd ? sAdd : sDel) + (long)b * S_DIM;
  float* o = (isAdd ? awAdd : awDel) + (long)b * S_DIM;
  __shared__ float red[256];
  const int tid = threadIdx.x;
  float mx = -1e30f;
  for (int i = tid; i < S_DIM; i += 256) mx = fmaxf(mx, s[i]);
  red[tid] = mx; __syncthreads();
  for (int st = 128; st > 0; st >>= 1) {
    if (tid < st) red[tid] = fmaxf(red[tid], red[tid + st]);
    __syncthreads();
  }
  mx = red[0]; __syncthreads();
  float sum = 0.f;
  for (int i = tid; i < S_DIM; i += 256) sum += __expf(s[i] - mx);
  red[tid] = sum; __syncthreads();
  for (int st = 128; st > 0; st >>= 1) {
    if (tid < st) red[tid] += red[tid + st];
    __syncthreads();
  }
  float inv = 1.f / red[0];
  for (int i = tid; i < S_DIM; i += 256) o[i] = __expf(s[i] - mx) * inv;
}

// cv_{del,add}[b,h] = sum_s aw[b,s] * enc[b,s,h]  ->  cvc[b, j, h]
// Each thread owns 4 contiguous columns (float4 -> global_load_b128).
__global__ __launch_bounds__(256) void context_kernel(
    const float* __restrict__ encDel, const float* __restrict__ encAdd,
    const float* __restrict__ awDel, const float* __restrict__ awAdd,
    float* __restrict__ cvc) {
  const int b = blockIdx.x & 31;
  const int j = blockIdx.x >> 5;
  const float* enc = (j ? encAdd : encDel) + (long)b * S_DIM * H_DIM;
  const float* aw  = (j ? awAdd : awDel) + (long)b * S_DIM;
  const int tid = threadIdx.x;
  float4 accv = {0.f, 0.f, 0.f, 0.f};
  for (int s = 0; s < S_DIM; ++s) {
    float w = aw[s];
    float4 rv = ((const float4*)(enc + (long)s * H_DIM))[tid];
    accv.x += w * rv.x;
    accv.y += w * rv.y;
    accv.z += w * rv.z;
    accv.w += w * rv.w;
  }
  ((float4*)(cvc + ((long)b * 2 + j) * H_DIM))[tid] = accv;
}

// softmax over the 2 common-attention scores; aw out; cv = aw0*cv_del + aw1*cv_add
__global__ __launch_bounds__(256) void combine_kernel(
    const float* __restrict__ sC, const float* __restrict__ cvc,
    float* __restrict__ awOut, float* __restrict__ cv) {
  const int b = blockIdx.x, tid = threadIdx.x;
  float s0 = sC[2 * b], s1 = sC[2 * b + 1];
  float mx = fmaxf(s0, s1);
  float e0 = __expf(s0 - mx), e1 = __expf(s1 - mx);
  float inv = 1.f / (e0 + e1);
  float a0 = e0 * inv, a1 = e1 * inv;
  if (tid == 0) { awOut[2 * b] = a0; awOut[2 * b + 1] = a1; }
  for (int h = tid; h < H_DIM; h += 256)
    cv[(long)b * H_DIM + h] = a0 * cvc[((long)b * 2) * H_DIM + h] +
                              a1 * cvc[((long)b * 2 + 1) * H_DIM + h];
}

// x_cat[b] = [ emb[inputs[b]] | cv[b] | h[b] ]   (B x 3072)
__global__ __launch_bounds__(256) void xcat_kernel(
    const int* __restrict__ inputs, const float* __restrict__ emb,
    const float* __restrict__ cv, const float* __restrict__ hin,
    float* __restrict__ xcat) {
  const int b = blockIdx.x, tid = threadIdx.x;
  long tok = inputs[b];
  for (int h = tid; h < H_DIM; h += 256) {
    xcat[(long)b * 3072 + h]        = emb[tok * H_DIM + h];
    xcat[(long)b * 3072 + 1024 + h] = cv[(long)b * H_DIM + h];
    xcat[(long)b * 3072 + 2048 + h] = hin[(long)b * H_DIM + h];
  }
}

// LSTM pointwise (PyTorch gate order i,f,g,o)
__global__ __launch_bounds__(256) void lstm_kernel(
    const float* __restrict__ gates, const float* __restrict__ cin,
    float* __restrict__ hOut, float* __restrict__ cOut) {
  const int idx = blockIdx.x * 256 + threadIdx.x;   // 0..B*H-1
  const int b = idx >> 10, n = idx & 1023;
  const float* g = gates + (long)b * 4096;
  float gi = g[n], gf = g[1024 + n], gg = g[2048 + n], go = g[3072 + n];
  float cn = fast_sigmoid(gf) * cin[idx] + fast_sigmoid(gi) * fast_tanh(gg);
  float hn = fast_sigmoid(go) * fast_tanh(cn);
  hOut[idx] = hn;
  cOut[idx] = cn;
}

// Online (max,sum) in one pass over logits, then one write pass.
__global__ __launch_bounds__(256) void logsoftmax_kernel(
    const float* __restrict__ logits, float* __restrict__ logp, int V) {
  const int b = blockIdx.x, tid = threadIdx.x;
  const float* x = logits + (long)b * V;
  float* o = logp + (long)b * V;
  __shared__ float rmx[256], rsum[256];
  float mx = -1e30f, sum = 0.f;
  for (int i = tid; i < V; i += 256) {
    float xv = x[i];
    if (xv > mx) {
      sum = sum * __expf(mx - xv) + 1.f;
      mx = xv;
    } else {
      sum += __expf(xv - mx);
    }
  }
  rmx[tid] = mx; rsum[tid] = sum; __syncthreads();
  for (int st = 128; st > 0; st >>= 1) {
    if (tid < st) {
      float m1 = rmx[tid], s1 = rsum[tid];
      float m2 = rmx[tid + st], s2 = rsum[tid + st];
      float mo = fmaxf(m1, m2);
      rmx[tid] = mo;
      rsum[tid] = s1 * __expf(m1 - mo) + s2 * __expf(m2 - mo);
    }
    __syncthreads();
  }
  float lse = rmx[0] + __logf(rsum[0]);
  for (int i = tid; i < V; i += 256) o[i] = x[i] - lse;
}

// ---------------------------------------------------------------------------
extern "C" void kernel_launch(void* const* d_in, const int* in_sizes, int n_in,
                              void* d_out, int out_size, void* d_ws, size_t ws_size,
                              hipStream_t stream) {
  (void)in_sizes; (void)n_in; (void)out_size; (void)ws_size;
  const int B = B_DIM, S = S_DIM, H = H_DIM, V = V_DIM;

  const int*   inputs = (const int*)d_in[0];
  const float* h      = (const float*)d_in[1];
  const float* c      = (const float*)d_in[2];
  const float* encDel = (const float*)d_in[3];
  const float* encAdd = (const float*)d_in[4];
  const float* emb    = (const float*)d_in[5];
  const float* W_h    = (const float*)d_in[6];
  const float* W_ed   = (const float*)d_in[7];
  const float* W_ea   = (const float*)d_in[8];
  const float* W_ec   = (const float*)d_in[9];
  const float* Vd_w   = (const float*)d_in[10];
  const float* Va_w   = (const float*)d_in[12];
  const float* Vc_w   = (const float*)d_in[14];
  const float* W_ih   = (const float*)d_in[16];
  const float* W_hh   = (const float*)d_in[17];
  const float* b_ih   = (const float*)d_in[18];
  const float* b_hh   = (const float*)d_in[19];
  const float* cls_w  = (const float*)d_in[20];
  const float* cls_b  = (const float*)d_in[21];
  // Vd_b / Va_b / Vc_b (d_in[11,13,15]) cancel in softmax -> unused.

  float* out      = (float*)d_out;
  float* outLogp  = out;                    // [B,V]
  float* outH     = outLogp + (long)B * V;  // [1,B,H]
  float* outC     = outH + B * H;           // [1,B,H]
  float* outAw    = outC + B * H;           // [B,2,1]
  float* outAwDel = outAw + B * 2;          // [B,S,1]
  float* outAwAdd = outAwDel + B * S;       // [B,S,1]

  float* ws    = (float*)d_ws;
  float* q     = ws;  ws += B * H;      // [B,H]
  float* sDel  = ws;  ws += B * S;      // [B,S]
  float* sAdd  = ws;  ws += B * S;
  float* cvc   = ws;  ws += B * 2 * H;  // [B,2,H]
  float* sC    = ws;  ws += 128;        // [B*2] (padded)
  float* cv    = ws;  ws += B * H;
  float* xcat  = ws;  ws += B * 3 * H;
  float* gates = ws;  ws += B * 4 * H;
  float* logits = ws;                   // [B,V]

  // 1. q = h @ W_h^T
  gemm32_kernel<<<H / 128, 128, 0, stream>>>(h, H, W_h, H, nullptr, 0,
                                             nullptr, nullptr, q, H);
  // 2. fused attention scores (GEMM + tanh + dot(V)), del & add
  fused_score_kernel<<<(B * S) / 256, 256, 0, stream>>>(encDel, W_ed, q, Vd_w,
                                                        sDel, B * S, S);
  fused_score_kernel<<<(B * S) / 256, 256, 0, stream>>>(encAdd, W_ea, q, Va_w,
                                                        sAdd, B * S, S);
  // 3. softmax over S -> aw_del / aw_add outputs
  softmaxS_kernel<<<64, 256, 0, stream>>>(sDel, sAdd, outAwDel, outAwAdd);
  // 4. context vectors -> cvc [B,2,H]
  context_kernel<<<64, 256, 0, stream>>>(encDel, encAdd, outAwDel, outAwAdd, cvc);
  // 5. common attention scores over cvc (64 rows, q shared per row-pair)
  fused_score_kernel<<<1, 256, 0, stream>>>(cvc, W_ec, q, Vc_w, sC, B * 2, 2);
  // 6. softmax over 2 + combine -> aw output, cv
  combine_kernel<<<B, 256, 0, stream>>>(sC, cvc, outAw, cv);
  // 7. x_cat = [emb | cv | h]
  xcat_kernel<<<B, 256, 0, stream>>>(inputs, emb, cv, h, xcat);
  // 8. gates = x_cat @ [W_ih|W_hh]^T + b_ih + b_hh  (K-split GEMM)
  gemm32_kernel<<<(4 * H) / 128, 128, 0, stream>>>(xcat, 3 * H, W_ih, 2 * H,
                                                   W_hh, H, b_ih, b_hh,
                                                   gates, 4 * H);
  // 9. LSTM pointwise -> h_new, c_new outputs
  lstm_kernel<<<(B * H) / 256, 256, 0, stream>>>(gates, c, outH, outC);
  // 10. logits = h_new @ cls_w^T + cls_b  (N-tail guarded)
  gemm32_kernel<<<(V + 127) / 128, 128, 0, stream>>>(outH, H, cls_w, H,
                                                     nullptr, 0, cls_b, nullptr,
                                                     logits, V);
  // 11. log_softmax over V -> logp output
  logsoftmax_kernel<<<B, 256, 0, stream>>>(logits, outLogp, V);
}
